// Obj2ObjGNN_78383153152533
// MI455X (gfx1250) — matrix operand order repaired and checked by reading, not compiled, not run on hardware
//
#include <hip/hip_runtime.h>
#include <hip/hip_bf16.h>
#include <math.h>

// ---- async global->LDS support detection ---------------------------------
#if __has_builtin(__builtin_amdgcn_global_load_async_to_lds_b128)
#define HAVE_ASYNC_LDS 1
#else
#define HAVE_ASYNC_LDS 0
#endif

// Problem constants (match reference)
#define Dd  256
#define Hh  256
#define N0n 4096
#define En  131072
#define TM  128       // rows per workgroup (8 waves x 16 rows)
#define MP_ITERS 3

#define XSTRIDE 264   // padded f16 row stride for X2 (bank-conflict-free A reads)
#define BSTRIDE 40    // padded f16 row stride for staged B slab
#define BSLAB   (256 * BSTRIDE)

typedef __attribute__((ext_vector_type(16))) _Float16 v16h;
typedef __attribute__((ext_vector_type(8)))  _Float16 v8h;
typedef __attribute__((ext_vector_type(8)))  float    v8f;
typedef int v4i_vs __attribute__((vector_size(16)));           // matches builtin param
typedef __attribute__((address_space(1))) v4i_vs* gptr_v4i;    // global AS pointer
typedef __attribute__((address_space(3))) v4i_vs* lptr_v4i;    // LDS AS pointer

// ---- WMMA fragment helpers (CDNA5 16x16x32 f16 layouts) -------------------
// A (16x32, MxK): lanes 0-15 -> M=lane, K elems {k0..k0+7, k0+16..k0+23}
//                 lanes16-31 -> M=lane-16, K elems {k0+8..k0+15, k0+24..k0+31}
static __device__ inline v16h wmma_a_from_f32(const float* __restrict__ rowp, int k0, int hh) {
  const float* p1 = rowp + k0 + 8 * hh;
  const float* p2 = p1 + 16;
  v16h r;
#pragma unroll
  for (int i = 0; i < 8; ++i) { r[i] = (_Float16)p1[i]; r[8 + i] = (_Float16)p2[i]; }
  return r;
}
static __device__ inline v16h wmma_a_from_f16(const _Float16* rowp, int k0, int hh) {
  v8h lo = *(const v8h*)(rowp + k0 + 8 * hh);
  v8h hi = *(const v8h*)(rowp + k0 + 8 * hh + 16);
  v16h r;
#pragma unroll
  for (int i = 0; i < 8; ++i) { r[i] = lo[i]; r[8 + i] = hi[i]; }
  return r;
}
// B (32x16, KxN): lanes 0-15 -> N=lane, K {k0..k0+7, k0+16..k0+23} ; lanes16-31 -> K {k0+8..k0+15, k0+24..k0+31}
static __device__ inline v16h lds_b(const _Float16* Bs, int n, int hh) {
  v8h lo = *(const v8h*)(Bs + n * BSTRIDE + 8 * hh);
  v8h hi = *(const v8h*)(Bs + n * BSTRIDE + 8 * hh + 16);
  v16h r;
#pragma unroll
  for (int i = 0; i < 8; ++i) { r[i] = lo[i]; r[8 + i] = hi[i]; }
  return r;
}

// ---- B-slab staging: thread n moves one 64B row slice of the K-slab -------
static __device__ inline void stage_slab(const _Float16* __restrict__ Wt, int K, int k0, int tid,
                                         _Float16* dst) {
#if HAVE_ASYNC_LDS
  const _Float16* g = Wt + (size_t)tid * K + k0;
  _Float16* l = dst + tid * BSTRIDE;
  __builtin_amdgcn_global_load_async_to_lds_b128((gptr_v4i)g, (lptr_v4i)l, 0,  0);
  __builtin_amdgcn_global_load_async_to_lds_b128((gptr_v4i)g, (lptr_v4i)l, 16, 0);
  __builtin_amdgcn_global_load_async_to_lds_b128((gptr_v4i)g, (lptr_v4i)l, 32, 0);
  __builtin_amdgcn_global_load_async_to_lds_b128((gptr_v4i)g, (lptr_v4i)l, 48, 0);
#else
  v16h g0 = *(const v16h*)(Wt + (size_t)tid * K + k0);
  v16h g1 = *(const v16h*)(Wt + (size_t)tid * K + k0 + 16);
  *(v16h*)(dst + tid * BSTRIDE) = g0;
  *(v16h*)(dst + tid * BSTRIDE + 16) = g1;
#endif
}
static __device__ inline void stage_wait() {
#if HAVE_ASYNC_LDS
#if __has_builtin(__builtin_amdgcn_s_wait_asynccnt)
  __builtin_amdgcn_s_wait_asynccnt(0);
#else
  asm volatile("s_wait_asynccnt 0x0" ::: "memory");
#endif
#endif
}

#define WMMA(acc, a, b) \
  (acc) = __builtin_amdgcn_wmma_f32_16x16x32_f16(false, (a), false, (b), (short)0, (acc), false, false)

// ---- utility kernels ------------------------------------------------------
__global__ void k_fill_zero(float* __restrict__ p, int n) {
  int i = blockIdx.x * blockDim.x + threadIdx.x;
  if (i < n) p[i] = 0.0f;
}
// W fp32 [K][N] -> Wt f16 [N][K]
__global__ void k_transpose_f16(const float* __restrict__ W, _Float16* __restrict__ Wt, int K, int N) {
  int idx = blockIdx.x * blockDim.x + threadIdx.x;
  if (idx >= K * N) return;
  int n = idx / K, k = idx - n * K;
  Wt[idx] = (_Float16)W[(size_t)k * N + n];
}

// ---- fused edge MLP + scatter-add ----------------------------------------
__global__ __launch_bounds__(256) void k_edge_mlp_scatter(
    const float* __restrict__ node0, const float* __restrict__ node1,
    const int* __restrict__ row, const int* __restrict__ col,
    const _Float16* __restrict__ W1, const float* __restrict__ b1,
    const _Float16* __restrict__ W2, const float* __restrict__ b2,
    const float* __restrict__ gam, const float* __restrict__ bet,
    const _Float16* __restrict__ W3, const float* __restrict__ b3,
    float* __restrict__ agg) {
  extern __shared__ _Float16 smem[];
  _Float16* X2 = smem;                  // [TM][XSTRIDE]
  _Float16* Bs = smem + TM * XSTRIDE;   // [2][BSLAB]

  const int tid   = threadIdx.x;
  const int lane  = tid & 31;
  const int wave  = tid >> 5;
  const int laneN = lane & 15;
  const int hh    = lane >> 4;
  const int r0    = wave * 16;
  const int eBase = blockIdx.x * TM;

  const float* a0 = node0 + (size_t)row[eBase + r0 + laneN] * Dd;
  const float* a1 = node1 + (size_t)col[eBase + r0 + laneN] * Dd;

  v8f zero = {};
  v8f acc[16];
#pragma unroll
  for (int t = 0; t < 16; ++t) acc[t] = zero;

  int cur = 0;

  // ---- layer 1: [TM x 512] @ [512 x 256] ----
  stage_slab(W1, 2 * Dd, 0, tid, Bs);
  stage_wait();
  __syncthreads();
  for (int ks = 0; ks < 16; ++ks) {
    if (ks + 1 < 16) stage_slab(W1, 2 * Dd, (ks + 1) * 32, tid, Bs + (cur ^ 1) * BSLAB);
    const float* arow = (ks < 8) ? a0 : a1;
    v16h a = wmma_a_from_f32(arow, (ks & 7) * 32, hh);
    const _Float16* bs = Bs + cur * BSLAB;
#pragma unroll
    for (int t = 0; t < 16; ++t) { v16h b = lds_b(bs, t * 16 + laneN, hh); WMMA(acc[t], a, b); }
    stage_wait();
    __syncthreads();
    cur ^= 1;
  }
  // bias + relu -> X2 (f16)
#pragma unroll
  for (int t = 0; t < 16; ++t) {
    const float bb = b1[t * 16 + laneN];
#pragma unroll
    for (int r = 0; r < 8; ++r) {
      float v = acc[t][r] + bb; v = v > 0.f ? v : 0.f;
      X2[(r0 + r + 8 * hh) * XSTRIDE + t * 16 + laneN] = (_Float16)v;
    }
    acc[t] = zero;
  }
  const _Float16* xrow = X2 + (r0 + laneN) * XSTRIDE;

  // ---- layer 2: [TM x 256] @ [256 x 256] ----
  stage_slab(W2, Hh, 0, tid, Bs + cur * BSLAB);
  stage_wait();
  __syncthreads();
  for (int ks = 0; ks < 8; ++ks) {
    if (ks + 1 < 8) stage_slab(W2, Hh, (ks + 1) * 32, tid, Bs + (cur ^ 1) * BSLAB);
    v16h a = wmma_a_from_f16(xrow, ks * 32, hh);
    const _Float16* bs = Bs + cur * BSLAB;
#pragma unroll
    for (int t = 0; t < 16; ++t) { v16h b = lds_b(bs, t * 16 + laneN, hh); WMMA(acc[t], a, b); }
    stage_wait();
    __syncthreads();
    cur ^= 1;
  }
  // bias + LayerNorm stats (rows live in 16-lane halves -> shfl_xor 1..8)
  float mu[8], rsg[8];
#pragma unroll
  for (int r = 0; r < 8; ++r) {
    float s = 0.f, q = 0.f;
#pragma unroll
    for (int t = 0; t < 16; ++t) { float v = acc[t][r] + b2[t * 16 + laneN]; s += v; q += v * v; }
#pragma unroll
    for (int m = 1; m < 16; m <<= 1) { s += __shfl_xor(s, m, 32); q += __shfl_xor(q, m, 32); }
    float mean = s * (1.f / (float)Hh);
    mu[r]  = mean;
    rsg[r] = rsqrtf(q * (1.f / (float)Hh) - mean * mean + 1e-5f);
  }
#pragma unroll
  for (int t = 0; t < 16; ++t) {
    const int c = t * 16 + laneN;
    const float bb = b2[c], gg = gam[c], ee = bet[c];
#pragma unroll
    for (int r = 0; r < 8; ++r) {
      float v = acc[t][r] + bb;
      v = (v - mu[r]) * rsg[r] * gg + ee;
      v = v > 0.f ? v : 0.f;
      X2[(r0 + r + 8 * hh) * XSTRIDE + c] = (_Float16)v;
    }
    acc[t] = zero;
  }
  // ---- layer 3 ----
  stage_slab(W3, Hh, 0, tid, Bs + cur * BSLAB);
  stage_wait();
  __syncthreads();
  for (int ks = 0; ks < 8; ++ks) {
    if (ks + 1 < 8) stage_slab(W3, Hh, (ks + 1) * 32, tid, Bs + (cur ^ 1) * BSLAB);
    v16h a = wmma_a_from_f16(xrow, ks * 32, hh);
    const _Float16* bs = Bs + cur * BSLAB;
#pragma unroll
    for (int t = 0; t < 16; ++t) { v16h b = lds_b(bs, t * 16 + laneN, hh); WMMA(acc[t], a, b); }
    stage_wait();
    __syncthreads();
    cur ^= 1;
  }
  // scatter-add edge_attr into agg[row]
  int ri[8];
#pragma unroll
  for (int r = 0; r < 8; ++r) ri[r] = row[eBase + r0 + r + 8 * hh];
#pragma unroll
  for (int t = 0; t < 16; ++t) {
    const int c = t * 16 + laneN;
    const float bb = b3[c];
#pragma unroll
    for (int r = 0; r < 8; ++r) atomicAdd(agg + (size_t)ri[r] * Hh + c, acc[t][r] + bb);
  }
}

// ---- fused node MLP + residual (in place on nodeCur) ----------------------
__global__ __launch_bounds__(256) void k_node_mlp_residual(
    float* __restrict__ nodeCur, const float* __restrict__ agg,
    const _Float16* __restrict__ W1, const float* __restrict__ b1,
    const _Float16* __restrict__ W2, const float* __restrict__ b2,
    const float* __restrict__ gam, const float* __restrict__ bet,
    const _Float16* __restrict__ W3, const float* __restrict__ b3) {
  extern __shared__ _Float16 smem[];
  _Float16* X2 = smem;
  _Float16* Bs = smem + TM * XSTRIDE;

  const int tid   = threadIdx.x;
  const int lane  = tid & 31;
  const int wave  = tid >> 5;
  const int laneN = lane & 15;
  const int hh    = lane >> 4;
  const int r0    = wave * 16;
  const int mBase = blockIdx.x * TM;

  const float* a0 = nodeCur + (size_t)(mBase + r0 + laneN) * Dd;
  const float* a1 = agg     + (size_t)(mBase + r0 + laneN) * Hh;

  v8f zero = {};
  v8f acc[16];
#pragma unroll
  for (int t = 0; t < 16; ++t) acc[t] = zero;

  int cur = 0;

  stage_slab(W1, Dd + Hh, 0, tid, Bs);
  stage_wait();
  __syncthreads();
  for (int ks = 0; ks < 16; ++ks) {
    if (ks + 1 < 16) stage_slab(W1, Dd + Hh, (ks + 1) * 32, tid, Bs + (cur ^ 1) * BSLAB);
    const float* arow = (ks < 8) ? a0 : a1;
    v16h a = wmma_a_from_f32(arow, (ks & 7) * 32, hh);
    const _Float16* bs = Bs + cur * BSLAB;
#pragma unroll
    for (int t = 0; t < 16; ++t) { v16h b = lds_b(bs, t * 16 + laneN, hh); WMMA(acc[t], a, b); }
    stage_wait();
    __syncthreads();
    cur ^= 1;
  }
#pragma unroll
  for (int t = 0; t < 16; ++t) {
    const float bb = b1[t * 16 + laneN];
#pragma unroll
    for (int r = 0; r < 8; ++r) {
      float v = acc[t][r] + bb; v = v > 0.f ? v : 0.f;
      X2[(r0 + r + 8 * hh) * XSTRIDE + t * 16 + laneN] = (_Float16)v;
    }
    acc[t] = zero;
  }
  const _Float16* xrow = X2 + (r0 + laneN) * XSTRIDE;

  stage_slab(W2, Hh, 0, tid, Bs + cur * BSLAB);
  stage_wait();
  __syncthreads();
  for (int ks = 0; ks < 8; ++ks) {
    if (ks + 1 < 8) stage_slab(W2, Hh, (ks + 1) * 32, tid, Bs + (cur ^ 1) * BSLAB);
    v16h a = wmma_a_from_f16(xrow, ks * 32, hh);
    const _Float16* bs = Bs + cur * BSLAB;
#pragma unroll
    for (int t = 0; t < 16; ++t) { v16h b = lds_b(bs, t * 16 + laneN, hh); WMMA(acc[t], a, b); }
    stage_wait();
    __syncthreads();
    cur ^= 1;
  }
  float mu[8], rsg[8];
#pragma unroll
  for (int r = 0; r < 8; ++r) {
    float s = 0.f, q = 0.f;
#pragma unroll
    for (int t = 0; t < 16; ++t) { float v = acc[t][r] + b2[t * 16 + laneN]; s += v; q += v * v; }
#pragma unroll
    for (int m = 1; m < 16; m <<= 1) { s += __shfl_xor(s, m, 32); q += __shfl_xor(q, m, 32); }
    float mean = s * (1.f / (float)Hh);
    mu[r]  = mean;
    rsg[r] = rsqrtf(q * (1.f / (float)Hh) - mean * mean + 1e-5f);
  }
#pragma unroll
  for (int t = 0; t < 16; ++t) {
    const int c = t * 16 + laneN;
    const float bb = b2[c], gg = gam[c], ee = bet[c];
#pragma unroll
    for (int r = 0; r < 8; ++r) {
      float v = acc[t][r] + bb;
      v = (v - mu[r]) * rsg[r] * gg + ee;
      v = v > 0.f ? v : 0.f;
      X2[(r0 + r + 8 * hh) * XSTRIDE + c] = (_Float16)v;
    }
    acc[t] = zero;
  }
  stage_slab(W3, Hh, 0, tid, Bs + cur * BSLAB);
  stage_wait();
  __syncthreads();
  for (int ks = 0; ks < 8; ++ks) {
    if (ks + 1 < 8) stage_slab(W3, Hh, (ks + 1) * 32, tid, Bs + (cur ^ 1) * BSLAB);
    v16h a = wmma_a_from_f16(xrow, ks * 32, hh);
    const _Float16* bs = Bs + cur * BSLAB;
#pragma unroll
    for (int t = 0; t < 16; ++t) { v16h b = lds_b(bs, t * 16 + laneN, hh); WMMA(acc[t], a, b); }
    stage_wait();
    __syncthreads();
    cur ^= 1;
  }
  // residual, in place (rows exclusive to this wave)
#pragma unroll
  for (int t = 0; t < 16; ++t) {
    const int c = t * 16 + laneN;
    const float bb = b3[c];
#pragma unroll
    for (int r = 0; r < 8; ++r) {
      size_t off = (size_t)(mBase + r0 + r + 8 * hh) * Dd + c;
      nodeCur[off] = acc[t][r] + bb + nodeCur[off];
    }
  }
}

// ---- fused edge predictor -------------------------------------------------
__global__ __launch_bounds__(256) void k_edge_predictor(
    const float* __restrict__ node0, const float* __restrict__ node1,
    const int* __restrict__ row, const int* __restrict__ col,
    const _Float16* __restrict__ W1, const float* __restrict__ b1,
    const float* __restrict__ w2, const float* __restrict__ b2,
    float* __restrict__ out) {
  extern __shared__ _Float16 smem[];
  _Float16* Bs = smem;  // [2][BSLAB]

  const int tid   = threadIdx.x;
  const int lane  = tid & 31;
  const int wave  = tid >> 5;
  const int laneN = lane & 15;
  const int hh    = lane >> 4;
  const int r0    = wave * 16;
  const int eBase = blockIdx.x * TM;

  const float* a0 = node0 + (size_t)row[eBase + r0 + laneN] * Dd;
  const float* a1 = node1 + (size_t)col[eBase + r0 + laneN] * Dd;

  v8f zero = {};
  v8f acc[16];
#pragma unroll
  for (int t = 0; t < 16; ++t) acc[t] = zero;

  int cur = 0;
  stage_slab(W1, 2 * Dd, 0, tid, Bs);
  stage_wait();
  __syncthreads();
  for (int ks = 0; ks < 16; ++ks) {
    if (ks + 1 < 16) stage_slab(W1, 2 * Dd, (ks + 1) * 32, tid, Bs + (cur ^ 1) * BSLAB);
    const float* arow = (ks < 8) ? a0 : a1;
    v16h a = wmma_a_from_f32(arow, (ks & 7) * 32, hh);
    const _Float16* bs = Bs + cur * BSLAB;
#pragma unroll
    for (int t = 0; t < 16; ++t) { v16h b = lds_b(bs, t * 16 + laneN, hh); WMMA(acc[t], a, b); }
    stage_wait();
    __syncthreads();
    cur ^= 1;
  }
  float s[8];
#pragma unroll
  for (int r = 0; r < 8; ++r) s[r] = 0.f;
#pragma unroll
  for (int t = 0; t < 16; ++t) {
    const int c = t * 16 + laneN;
    const float bb = b1[c], ww = w2[c];
#pragma unroll
    for (int r = 0; r < 8; ++r) {
      float v = acc[t][r] + bb; v = v > 0.f ? v : 0.f;
      s[r] += v * ww;
    }
  }
#pragma unroll
  for (int r = 0; r < 8; ++r) {
#pragma unroll
    for (int m = 1; m < 16; m <<= 1) s[r] += __shfl_xor(s[r], m, 32);
  }
  const float pb = b2[0];
  if (laneN == 0) {
#pragma unroll
    for (int r = 0; r < 8; ++r) {
      float v = s[r] + pb;
      out[eBase + r0 + r + 8 * hh] = 1.f / (1.f + expf(-v));
    }
  }
}

// ---- host orchestration ---------------------------------------------------
extern "C" void kernel_launch(void* const* d_in, const int* in_sizes, int n_in,
                              void* d_out, int out_size, void* d_ws, size_t ws_size,
                              hipStream_t stream) {
  (void)in_sizes; (void)n_in; (void)out_size; (void)ws_size;
  const float* node0 = (const float*)d_in[0];
  const float* node1 = (const float*)d_in[1];
  const int*   row   = (const int*)d_in[2];
  const int*   col   = (const int*)d_in[3];
  const float* e_w1 = (const float*)d_in[4];  const float* e_b1 = (const float*)d_in[5];
  const float* e_w2 = (const float*)d_in[6];  const float* e_b2 = (const float*)d_in[7];
  const float* e_g  = (const float*)d_in[8];  const float* e_be = (const float*)d_in[9];
  const float* e_w3 = (const float*)d_in[10]; const float* e_b3 = (const float*)d_in[11];
  const float* n_w1 = (const float*)d_in[12]; const float* n_b1 = (const float*)d_in[13];
  const float* n_w2 = (const float*)d_in[14]; const float* n_b2 = (const float*)d_in[15];
  const float* n_g  = (const float*)d_in[16]; const float* n_be = (const float*)d_in[17];
  const float* n_w3 = (const float*)d_in[18]; const float* n_b3 = (const float*)d_in[19];
  const float* p_w1 = (const float*)d_in[20]; const float* p_b1 = (const float*)d_in[21];
  const float* p_w2 = (const float*)d_in[22]; const float* p_b2 = (const float*)d_in[23];

  char* ws = (char*)d_ws;
  size_t off = 0;
  auto a16 = [&](size_t elems) -> _Float16* { _Float16* p = (_Float16*)(ws + off); off += elems * sizeof(_Float16); return p; };
  _Float16* eW1t = a16((size_t)(2 * Dd) * Hh);
  _Float16* eW2t = a16((size_t)Hh * Hh);
  _Float16* eW3t = a16((size_t)Hh * Hh);
  _Float16* nW1t = a16((size_t)(Dd + Hh) * Hh);
  _Float16* nW2t = a16((size_t)Hh * Hh);
  _Float16* nW3t = a16((size_t)Hh * Dd);
  _Float16* pW1t = a16((size_t)(2 * Dd) * Hh);
  off = (off + 255) & ~(size_t)255;
  float* agg = (float*)(ws + off); off += (size_t)N0n * Hh * sizeof(float);
  float* nodeCur = (float*)(ws + off); off += (size_t)N0n * Dd * sizeof(float);

  const int TP = 256;
  auto tgrid = [&](int K, int N) { return (K * N + TP - 1) / TP; };
  k_transpose_f16<<<tgrid(2 * Dd, Hh), TP, 0, stream>>>(e_w1, eW1t, 2 * Dd, Hh);
  k_transpose_f16<<<tgrid(Hh, Hh),     TP, 0, stream>>>(e_w2, eW2t, Hh, Hh);
  k_transpose_f16<<<tgrid(Hh, Hh),     TP, 0, stream>>>(e_w3, eW3t, Hh, Hh);
  k_transpose_f16<<<tgrid(Dd + Hh, Hh),TP, 0, stream>>>(n_w1, nW1t, Dd + Hh, Hh);
  k_transpose_f16<<<tgrid(Hh, Hh),     TP, 0, stream>>>(n_w2, nW2t, Hh, Hh);
  k_transpose_f16<<<tgrid(Hh, Dd),     TP, 0, stream>>>(n_w3, nW3t, Hh, Dd);
  k_transpose_f16<<<tgrid(2 * Dd, Hh), TP, 0, stream>>>(p_w1, pW1t, 2 * Dd, Hh);

  (void)hipMemcpyAsync(nodeCur, node0, (size_t)N0n * Dd * sizeof(float),
                       hipMemcpyDeviceToDevice, stream);

  const size_t ldsMain = (size_t)(TM * XSTRIDE + 2 * BSLAB) * sizeof(_Float16); // ~106 KB
  const size_t ldsPred = (size_t)(2 * BSLAB) * sizeof(_Float16);                // ~40 KB

  for (int it = 0; it < MP_ITERS; ++it) {
    k_fill_zero<<<(N0n * Hh + TP - 1) / TP, TP, 0, stream>>>(agg, N0n * Hh);
    k_edge_mlp_scatter<<<En / TM, 256, ldsMain, stream>>>(nodeCur, node1, row, col,
        eW1t, e_b1, eW2t, e_b2, e_g, e_be, eW3t, e_b3, agg);
    k_node_mlp_residual<<<N0n / TM, 256, ldsMain, stream>>>(nodeCur, agg,
        nW1t, n_b1, nW2t, n_b2, n_g, n_be, nW3t, n_b3);
  }
  k_edge_predictor<<<En / TM, 256, ldsPred, stream>>>(nodeCur, node1, row, col,
      pW1t, p_b1, p_w2, p_b2, (float*)d_out);
}
